// complexSelfMHA_47304769798717
// MI455X (gfx1250) — matrix operand order repaired and checked by reading, not compiled, not run on hardware
//
#include <hip/hip_runtime.h>

typedef __attribute__((ext_vector_type(16))) _Float16 v16h;
typedef __attribute__((ext_vector_type(8)))  float    v8f;

#define EMB   1024
#define HEADS 16
#define HD    64
#define BATCH 4
#define SEQ   1024
#define MTOT  (BATCH * SEQ)   // 4096 rows when (b,s) flattened

union FragU { v16h h; uint4 u[2]; };

// ---- WMMA f16 16x16x32, f32 accumulate -------------------------------------
__device__ __forceinline__ v8f wmma16(v16h a, v16h b, v8f c) {
  return __builtin_amdgcn_wmma_f32_16x16x32_f16(
      /*neg_a=*/false, a, /*neg_b=*/false, b,
      /*c_mod=*/(short)0, c, /*reuse_a=*/false, /*reuse_b=*/false);
}

// A operand (16x32 f16): lane = row (lane&15); K = (lane>>4)*8 + {0..7,16..23}
__device__ __forceinline__ v16h load_fragA(const _Float16* base, int ld, int r16, int kk) {
  int lane = threadIdx.x & 31;
  const _Float16* p = base + (size_t)(r16 + (lane & 15)) * ld + kk + ((lane >> 4) << 3);
  FragU f;
  f.u[0] = *(const uint4*)p;
  f.u[1] = *(const uint4*)(p + 16);
  return f.h;
}

// B operand (32x16 f16) from "Bt" memory laid out [n][k]:
// lane = col n (lane&15); K = (lane>>4)*8 + {0..7,16..23}
__device__ __forceinline__ v16h load_fragB(const _Float16* base, int ld, int n16, int kk) {
  int lane = threadIdx.x & 31;
  const _Float16* p = base + (size_t)(n16 + (lane & 15)) * ld + kk + ((lane >> 4) << 3);
  FragU f;
  f.u[0] = *(const uint4*)p;
  f.u[1] = *(const uint4*)(p + 16);
  return f.h;
}

// A operand built from f32 LDS scores (softmax probs), convert on the fly
__device__ __forceinline__ v16h load_fragA_lds(const float* sc, int ld, int kk) {
  int lane = threadIdx.x & 31;
  const float* p = sc + (size_t)(lane & 15) * ld + kk + ((lane >> 4) << 3);
  v16h o;
#pragma unroll
  for (int i = 0; i < 8; ++i) {
    o[i]     = (_Float16)p[i];
    o[i + 8] = (_Float16)p[i + 16];
  }
  return o;
}

// ---- elementwise prep kernels ----------------------------------------------
__global__ void __launch_bounds__(256) cvt_f32_f16(const float* __restrict__ x,
                                                   _Float16* __restrict__ y, int n) {
  int i = blockIdx.x * 256 + threadIdx.x;
  if (i < n) y[i] = (_Float16)x[i];
}

// Wt[n][k] = W[k][n], f32 -> f16   (1024x1024)
__global__ void __launch_bounds__(256) wtrans(const float* __restrict__ W,
                                              _Float16* __restrict__ Wt) {
  int i = blockIdx.x * 256 + threadIdx.x;   // i = n*1024 + k
  int n = i >> 10, k = i & 1023;
  Wt[i] = (_Float16)W[(size_t)k * EMB + n];
}

// I16[b][h][s][d] = (f16) Im[b][s][h*64+d]
__global__ void __launch_bounds__(256) imsplit(const float* __restrict__ Im,
                                               _Float16* __restrict__ I16) {
  int i = blockIdx.x * 256 + threadIdx.x;   // i = (((b*16+h)<<10)+s)*64 + d
  int d  = i & 63;
  int s  = (i >> 6) & 1023;
  int bh = i >> 16;
  int b = bh >> 4, h = bh & 15;
  I16[i] = (_Float16)Im[((size_t)((b << 10) + s)) * EMB + h * HD + d];
}

// ---- generalized WMMA GEMM:  C(4096x1024) = A16(4096x1024) * Bt^T ----------
// Each wave: 32x64 strip (2 A frags shared across 4 B frags, 8 accumulators),
// software-pipelined: next k-step fragments are loaded before current WMMAs.
// mode 0: head-split, +bias +pos, store f16 [b][h][s][d]            (Q, K)
// mode 1: head-split, +bias +pos, store f16 transposed [b][h][d][s] (V)
// mode 2: +bias, store f32 row-major (final output)
__global__ void __launch_bounds__(256) gemm_wmma(
    const _Float16* __restrict__ A, const _Float16* __restrict__ Bt,
    const float* __restrict__ bias, const float* __restrict__ pos,
    _Float16* __restrict__ oh, float* __restrict__ of, int mode) {
  int wave = threadIdx.x >> 5, lane = threadIdx.x & 31;
  int strip = blockIdx.x * 8 + wave;        // 2048 strips of 32 rows x 64 cols
  int m0 = (strip >> 4) << 5;               // 128 row-strips
  int n0 = (strip & 15) << 6;               // 16  col-strips

  v8f acc[2][4] = {};

  // prologue: fragments for kk = 0
  v16h a0 = load_fragA(A, EMB, m0,      0);
  v16h a1 = load_fragA(A, EMB, m0 + 16, 0);
  v16h b0 = load_fragB(Bt, EMB, n0,      0);
  v16h b1 = load_fragB(Bt, EMB, n0 + 16, 0);
  v16h b2 = load_fragB(Bt, EMB, n0 + 32, 0);
  v16h b3 = load_fragB(Bt, EMB, n0 + 48, 0);

  for (int kk = 0; kk < EMB - 32; kk += 32) {
    int kn = kk + 32;
    // issue next k-step loads first so they overlap the WMMAs below
    v16h a0n = load_fragA(A, EMB, m0,      kn);
    v16h a1n = load_fragA(A, EMB, m0 + 16, kn);
    v16h b0n = load_fragB(Bt, EMB, n0,      kn);
    v16h b1n = load_fragB(Bt, EMB, n0 + 16, kn);
    v16h b2n = load_fragB(Bt, EMB, n0 + 32, kn);
    v16h b3n = load_fragB(Bt, EMB, n0 + 48, kn);

    acc[0][0] = wmma16(a0, b0, acc[0][0]);
    acc[1][0] = wmma16(a1, b0, acc[1][0]);
    acc[0][1] = wmma16(a0, b1, acc[0][1]);
    acc[1][1] = wmma16(a1, b1, acc[1][1]);
    acc[0][2] = wmma16(a0, b2, acc[0][2]);
    acc[1][2] = wmma16(a1, b2, acc[1][2]);
    acc[0][3] = wmma16(a0, b3, acc[0][3]);
    acc[1][3] = wmma16(a1, b3, acc[1][3]);

    a0 = a0n; a1 = a1n; b0 = b0n; b1 = b1n; b2 = b2n; b3 = b3n;
  }
  // epilogue k-step
  acc[0][0] = wmma16(a0, b0, acc[0][0]);
  acc[1][0] = wmma16(a1, b0, acc[1][0]);
  acc[0][1] = wmma16(a0, b1, acc[0][1]);
  acc[1][1] = wmma16(a1, b1, acc[1][1]);
  acc[0][2] = wmma16(a0, b2, acc[0][2]);
  acc[1][2] = wmma16(a1, b2, acc[1][2]);
  acc[0][3] = wmma16(a0, b3, acc[0][3]);
  acc[1][3] = wmma16(a1, b3, acc[1][3]);

  int ml0 = (lane >> 4) << 3, nl = lane & 15;
#pragma unroll
  for (int i = 0; i < 2; ++i) {
#pragma unroll
    for (int j = 0; j < 4; ++j) {
#pragma unroll
      for (int r = 0; r < 8; ++r) {
        int m = m0 + i * 16 + ml0 + r;
        int n = n0 + j * 16 + nl;
        float v = acc[i][j][r] + bias[n];
        if (mode == 2) {
          of[(size_t)m * EMB + n] = v;
        } else {
          int b = m >> 10, s = m & 1023, h = n >> 6, d = n & 63;
          v += pos[((size_t)((b << 10) + s)) * HD + d];
          if (mode == 0)
            oh[(((size_t)(b * HEADS + h) << 10) + s) * HD + d] = (_Float16)v;
          else
            oh[((size_t)(b * HEADS + h) * HD + d) * SEQ + s] = (_Float16)v;
        }
      }
    }
  }
}

// ---- attention: one block = one (b,h) and 16 query rows --------------------
// scores (16 x 1024 f32) live in LDS; softmax via wave32 shuffles; P*V WMMA.
__global__ void __launch_bounds__(256) attn_wmma(
    const _Float16* __restrict__ Q, const _Float16* __restrict__ K,
    const _Float16* __restrict__ Vt, const _Float16* __restrict__ I,
    _Float16* __restrict__ Aout) {
  extern __shared__ float sc[];             // 16 * 1024 f32 = 64 KB
  int blk = blockIdx.x;                     // 4096 = 64 (b,h) * 64 q-tiles
  int qb  = blk & 63;
  int bh  = blk >> 6;
  int b = bh >> 4, h = bh & 15;
  int q0 = qb << 4;
  int wave = threadIdx.x >> 5, lane = threadIdx.x & 31;

  const _Float16* Qbh = Q  + (size_t)bh * SEQ * HD;
  const _Float16* Kbh = K  + (size_t)bh * SEQ * HD;
  const _Float16* Ibh = I  + (size_t)bh * SEQ * HD;
  const _Float16* Vbh = Vt + (size_t)bh * HD * SEQ;   // [d][s]

  // query-side fragments (reused across all 64 key tiles)
  v16h qa0 = load_fragA(Qbh, HD, q0, 0);
  v16h qa1 = load_fragA(Qbh, HD, q0, 32);
  v16h ia0 = load_fragA(Ibh, HD, q0, 0);
  v16h ia1 = load_fragA(Ibh, HD, q0, 32);

  // phase 1: score tiles -> LDS (8 key tiles per wave), pipelined
  int ml0 = (lane >> 4) << 3, nl = lane & 15;
  {
    int k0 = wave << 4;
    v16h kb0 = load_fragB(Kbh, HD, k0, 0);
    v16h kb1 = load_fragB(Kbh, HD, k0, 32);
    v16h ib0 = load_fragB(Ibh, HD, k0, 0);
    v16h ib1 = load_fragB(Ibh, HD, k0, 32);
    for (int t = wave; t < SEQ / 16; t += 8) {
      int kc = t << 4;
      v16h kb0n = kb0, kb1n = kb1, ib0n = ib0, ib1n = ib1;
      if (t + 8 < SEQ / 16) {                 // uniform per wave
        int kn = (t + 8) << 4;
        kb0n = load_fragB(Kbh, HD, kn, 0);
        kb1n = load_fragB(Kbh, HD, kn, 32);
        ib0n = load_fragB(Ibh, HD, kn, 0);
        ib1n = load_fragB(Ibh, HD, kn, 32);
      }
      v8f acc = {};
      acc = wmma16(qa0, kb0, acc);
      acc = wmma16(qa1, kb1, acc);
      acc = wmma16(ia0, ib0, acc);
      acc = wmma16(ia1, ib1, acc);
#pragma unroll
      for (int r = 0; r < 8; ++r)
        sc[(ml0 + r) * SEQ + kc + nl] = acc[r] * 0.03125f;   // 1/sqrt(1024)
      kb0 = kb0n; kb1 = kb1n; ib0 = ib0n; ib1 = ib1n;
    }
  }
  __syncthreads();

  // phase 2: softmax, 2 rows per wave, lane-parallel + shuffle reduce
#pragma unroll
  for (int rr = 0; rr < 2; ++rr) {
    float* p = sc + (size_t)(wave * 2 + rr) * SEQ;
    float mx = -3.402823466e38f;
    for (int c = lane; c < SEQ; c += 32) mx = fmaxf(mx, p[c]);
#pragma unroll
    for (int off = 16; off; off >>= 1) mx = fmaxf(mx, __shfl_xor(mx, off, 32));
    float sum = 0.f;
    for (int c = lane; c < SEQ; c += 32) {
      float e = __expf(p[c] - mx);
      p[c] = e;
      sum += e;
    }
#pragma unroll
    for (int off = 16; off; off >>= 1) sum += __shfl_xor(sum, off, 32);
    float inv = 1.0f / sum;
    for (int c = lane; c < SEQ; c += 32) p[c] *= inv;
  }
  __syncthreads();

  // phase 3: out(16 x 64) = P(16 x 1024) * V(1024 x 64); waves 0..3 take d-tiles
  if (wave < 4) {
    int d0 = wave << 4;
    v8f acc = {};
    v16h vb = load_fragB(Vbh, SEQ, d0, 0);
    for (int kk = 0; kk < SEQ; kk += 32) {
      v16h vbn = vb;
      if (kk + 32 < SEQ)
        vbn = load_fragB(Vbh, SEQ, d0, kk + 32);   // overlap with LDS reads + WMMA
      v16h pa = load_fragA_lds(sc, SEQ, kk);
      acc = wmma16(pa, vb, acc);
      vb = vbn;
    }
#pragma unroll
    for (int r = 0; r < 8; ++r) {
      int s = q0 + ml0 + r;
      int e = h * HD + d0 + nl;
      Aout[((size_t)((b << 10) + s)) * EMB + e] = (_Float16)acc[r];
    }
  }
}

// ---- host side -------------------------------------------------------------
extern "C" void kernel_launch(void* const* d_in, const int* in_sizes, int n_in,
                              void* d_out, int out_size, void* d_ws, size_t ws_size,
                              hipStream_t stream) {
  const float* Re  = (const float*)d_in[0];
  const float* Im  = (const float*)d_in[1];
  const float* pos = (const float*)d_in[2];
  const float* Wq  = (const float*)d_in[3];
  const float* bq  = (const float*)d_in[4];
  const float* Wk  = (const float*)d_in[5];
  const float* bk  = (const float*)d_in[6];
  const float* Wv  = (const float*)d_in[7];
  const float* bv  = (const float*)d_in[8];
  const float* Wo  = (const float*)d_in[9];
  const float* bo  = (const float*)d_in[10];
  float* out = (float*)d_out;

  const size_t NE = (size_t)MTOT * EMB;     // 4,194,304 elements
  const size_t WE = (size_t)EMB * EMB;      // 1,048,576 elements

  char* ws = (char*)d_ws;
  _Float16* Re16   = (_Float16*)ws; ws += NE * 2;
  _Float16* Wqt    = (_Float16*)ws; ws += WE * 2;
  _Float16* Wkt    = (_Float16*)ws; ws += WE * 2;
  _Float16* Wvt    = (_Float16*)ws; ws += WE * 2;
  _Float16* Wot    = (_Float16*)ws; ws += WE * 2;
  _Float16* Q16    = (_Float16*)ws; ws += NE * 2;   // [b][h][s][d]
  _Float16* K16    = (_Float16*)ws; ws += NE * 2;   // [b][h][s][d]
  _Float16* Vt16   = (_Float16*)ws; ws += NE * 2;   // [b][h][d][s]
  _Float16* I16    = (_Float16*)ws; ws += NE * 2;   // [b][h][s][d]
  _Float16* Aout16 = (_Float16*)ws; ws += NE * 2;   // [b][s][e]

  dim3 blk(256);

  cvt_f32_f16<<<(int)(NE / 256), blk, 0, stream>>>(Re, Re16, (int)NE);
  wtrans<<<(int)(WE / 256), blk, 0, stream>>>(Wq, Wqt);
  wtrans<<<(int)(WE / 256), blk, 0, stream>>>(Wk, Wkt);
  wtrans<<<(int)(WE / 256), blk, 0, stream>>>(Wv, Wvt);
  wtrans<<<(int)(WE / 256), blk, 0, stream>>>(Wo, Wot);
  imsplit<<<(int)(NE / 256), blk, 0, stream>>>(Im, I16);

  // projections: 2048 strips / 8 waves per block = 256 blocks
  gemm_wmma<<<256, blk, 0, stream>>>(Re16, Wqt, bq, pos, Q16,  nullptr, 0);
  gemm_wmma<<<256, blk, 0, stream>>>(Re16, Wkt, bk, pos, K16,  nullptr, 0);
  gemm_wmma<<<256, blk, 0, stream>>>(Re16, Wvt, bv, pos, Vt16, nullptr, 1);

  // attention: 64 (b,h) * 64 query tiles; 64 KB dynamic LDS for the score rows
  attn_wmma<<<BATCH * HEADS * (SEQ / 16), blk, 16 * SEQ * sizeof(float), stream>>>(
      Q16, K16, Vt16, I16, Aout16);

  // output projection -> f32
  gemm_wmma<<<256, blk, 0, stream>>>(Aout16, Wot, bo, nullptr, nullptr, out, 2);
}
